// cell_CELL_31439160607301
// MI455X (gfx1250) — compile-verified, matrix operand order checked by reading, not run on hardware
//
#include <hip/hip_runtime.h>
#include <hip/hip_bf16.h>
#include <cmath>

typedef __attribute__((ext_vector_type(16))) _Float16 v16h;
typedef __attribute__((ext_vector_type(8)))  _Float16 v8h;
typedef __attribute__((ext_vector_type(8)))  float    v8f;
typedef __attribute__((ext_vector_type(4)))  float    v4f;

#define Bsz 64
#define Ssz 1024
#define Dsz 512
#define NT  (Dsz/16)        // 32 column tiles of 16
#define KT  (Dsz/32)        // 16 K tiles of 32
#define HS  (Dsz + 8)       // padded LDS row stride (halves) -> conflict-free A loads
#define WTILE_H (KT*32*16)  // 8192 halves = 16 KB packed-W per column tile

// ---------------------------------------------------------------------------
// Pack W[N=512][K=512] (row-major f32) into WMMA-B fragment-major f16.
// B operand (K x N): lane = {g=lane>>4, c=lane&15}, element e maps to
//   K = kk*32 + 8g + (e<8 ? e : e+8),  N = nt*16 + c
// Wp[((nt*KT + kk)*32 + lane)*16 + e]  -> each nt tile is 16 KB contiguous.
// ---------------------------------------------------------------------------
__global__ void pack_w_kernel(const float* __restrict__ W, _Float16* __restrict__ Wp) {
    int idx  = blockIdx.x * blockDim.x + threadIdx.x;   // 262144 total
    int e    = idx & 15;
    int lane = (idx >> 4) & 31;
    int kk   = (idx >> 9) & (KT - 1);
    int nt   = idx >> 13;
    int g = lane >> 4, c = lane & 15;
    int k = kk * 32 + 8 * g + (e < 8 ? e : e + 8);
    Wp[idx] = (_Float16)W[(size_t)(nt * 16 + c) * Dsz + k];
}

// A fragment (16x32 f16) from row-major f32 [rows][512], converted on the fly.
__device__ __forceinline__ v16h load_a_frag_f32(const float* __restrict__ base, int k0, int lane) {
    int g = lane >> 4, m = lane & 15;
    const float* p = base + (size_t)m * Dsz + (k0 + 8 * g);
    v4f a0 = *(const v4f*)(p +  0);
    v4f a1 = *(const v4f*)(p +  4);
    v4f a2 = *(const v4f*)(p + 16);
    v4f a3 = *(const v4f*)(p + 20);
    v16h r;
#pragma unroll
    for (int i = 0; i < 4; ++i) {
        r[i]      = (_Float16)a0[i];
        r[4 + i]  = (_Float16)a1[i];
        r[8 + i]  = (_Float16)a2[i];
        r[12 + i] = (_Float16)a3[i];
    }
    return r;
}

__device__ __forceinline__ v16h load_b_frag(const _Float16* __restrict__ Wp, int nt, int kk, int lane) {
    return *(const v16h*)(Wp + (((size_t)(nt * KT + kk)) * 32 + lane) * 16);  // 32B aligned
}

__device__ __forceinline__ v16h lds_b_frag(const _Float16* tile, int kk, int lane) {
    return *(const v16h*)(tile + (kk * 32 + lane) * 16);                      // 2x ds_load_b128
}

// ---------------------------------------------------------------------------
// Phase 1: xt_all = x @ W_ihT + b_ih   (M = 65536, N = K = 512)
// 512 blocks x 8 waves; wave owns one 16-row tile, A fragments pinned in
// registers. Per-nt 16 KB W tile staged once per block into double-buffered
// LDS (global loads for nt+1 issued before the WMMA chain of nt). The LDS
// B fragment is software-pipelined one kk ahead so each WMMA waits on the
// *previous* iteration's ds loads, not the ones just issued.
// ---------------------------------------------------------------------------
__global__ void __launch_bounds__(256)
proj_kernel(const float* __restrict__ x, const _Float16* __restrict__ Wp,
            const float* __restrict__ b_ih, float* __restrict__ out) {
    __shared__ __align__(32) _Float16 wtile[2][WTILE_H];   // 2 x 16 KB

    int tid  = threadIdx.x;
    int lane = tid & 31;
    int wave = tid >> 5;                         // 0..7
    int row0 = (blockIdx.x * 8 + wave) * 16;
    const float* abase = x + (size_t)row0 * Dsz;

    v16h a[KT];
#pragma unroll
    for (int kk = 0; kk < KT; ++kk) a[kk] = load_a_frag_f32(abase, kk * 32, lane);

    // stage tile 0 (each thread copies 64B, fully coalesced)
    {
        const v8h* src = (const v8h*)Wp + tid * 4;
        v8h r0 = src[0], r1 = src[1], r2 = src[2], r3 = src[3];
        v8h* dst = (v8h*)wtile[0] + tid * 4;
        dst[0] = r0; dst[1] = r1; dst[2] = r2; dst[3] = r3;
    }
    __syncthreads();

    int g = lane >> 4, c = lane & 15;
    for (int nt = 0; nt < NT; ++nt) {
        int buf = nt & 1;
        int ntn = (nt + 1) & (NT - 1);
        // issue global loads for next W tile (consumed after the WMMA chain)
        const v8h* src = (const v8h*)(Wp + (size_t)ntn * WTILE_H) + tid * 4;
        v8h r0 = src[0], r1 = src[1], r2 = src[2], r3 = src[3];

        const _Float16* tile = wtile[buf];
        v8f acc = {};
        v16h bcur = lds_b_frag(tile, 0, lane);
#pragma unroll
        for (int kk = 0; kk < KT; ++kk) {
            v16h bnext;                          // pipeline LDS B one kk ahead
            if (kk + 1 < KT) bnext = lds_b_frag(tile, kk + 1, lane);
            acc = __builtin_amdgcn_wmma_f32_16x16x32_f16(false, a[kk], false, bcur,
                                                         (short)0, acc, false, false);
            if (kk + 1 < KT) bcur = bnext;
        }

        // write next tile to LDS (global latency now fully covered)
        v8h* dst = (v8h*)wtile[buf ^ 1] + tid * 4;
        dst[0] = r0; dst[1] = r1; dst[2] = r2; dst[3] = r3;

        int col = nt * 16 + c;
        float bias = b_ih[col];
#pragma unroll
        for (int r = 0; r < 8; ++r)
            out[(size_t)(row0 + 8 * g + r) * Dsz + col] = acc[r] + bias;

        __syncthreads();
    }
}

// ---------------------------------------------------------------------------
// Phase 2: recurrent scan. Batch rows independent -> 4 workgroups of 16 batch
// rows, intra-workgroup barrier per timestep, h double-buffered f16 in LDS.
// A (LDS) and B (global) fragments both software-pipelined one kk ahead;
// xt loads hoisted to the top of each step; next-step xt prefetched; final
// stores nontemporal.
// ---------------------------------------------------------------------------
__global__ void __launch_bounds__(512)
scan_kernel(const _Float16* __restrict__ Wp, const float* __restrict__ b_hh,
            float* __restrict__ out, float* __restrict__ last) {
    __shared__ __align__(32) _Float16 hbuf[2][16 * HS];

    int tid  = threadIdx.x;
    int lane = tid & 31;
    int wave = tid >> 5;                 // 0..15
    int brow0 = blockIdx.x * 16;

    for (int i = tid; i < 16 * HS; i += 512) hbuf[0][i] = (_Float16)0.f;
    __syncthreads();

    int n0 = 2 * wave, n1 = n0 + 1;
    int g = lane >> 4, c = lane & 15;
    int col0 = n0 * 16 + c, col1 = n1 * 16 + c;
    float bias0 = b_hh[col0];
    float bias1 = b_hh[col1];

    size_t rowbase[8];
#pragma unroll
    for (int r = 0; r < 8; ++r)
        rowbase[r] = (size_t)(brow0 + 8 * g + r) * Ssz * Dsz;

    for (int t = 0; t < Ssz; ++t) {
        int cur = t & 1, nxt = cur ^ 1;
        const _Float16* hb = hbuf[cur] + (size_t)c * HS + 8 * g;

        // hoist xt loads for this step; prefetch next step's xt into cache
        float xt0[8], xt1[8];
#pragma unroll
        for (int r = 0; r < 8; ++r) {
            size_t base = rowbase[r] + (size_t)t * Dsz;
            xt0[r] = out[base + col0];
            xt1[r] = out[base + col1];
            if (t + 1 < Ssz) {
                __builtin_prefetch(&out[base + Dsz + col0], 0, 3);
                __builtin_prefetch(&out[base + Dsz + col1], 0, 3);
            }
        }

        v8f acc0 = {}, acc1 = {};
        v16h bf0 = load_b_frag(Wp, n0, 0, lane);
        v16h bf1 = load_b_frag(Wp, n1, 0, lane);
        v8h  alo = *(const v8h*)(hb);
        v8h  ahi = *(const v8h*)(hb + 16);
#pragma unroll
        for (int kk = 0; kk < KT; ++kk) {
            v16h av;
#pragma unroll
            for (int i = 0; i < 8; ++i) { av[i] = alo[i]; av[8 + i] = ahi[i]; }

            v8h nlo, nhi;                       // pipeline LDS A one kk ahead
            v16h nb0, nb1;                      // pipeline global B one kk ahead
            if (kk + 1 < KT) {
                const _Float16* p = hb + (kk + 1) * 32;
                nlo = *(const v8h*)(p);
                nhi = *(const v8h*)(p + 16);
                nb0 = load_b_frag(Wp, n0, kk + 1, lane);
                nb1 = load_b_frag(Wp, n1, kk + 1, lane);
            }
            acc0 = __builtin_amdgcn_wmma_f32_16x16x32_f16(false, av, false, bf0,
                                                          (short)0, acc0, false, false);
            acc1 = __builtin_amdgcn_wmma_f32_16x16x32_f16(false, av, false, bf1,
                                                          (short)0, acc1, false, false);
            if (kk + 1 < KT) { alo = nlo; ahi = nhi; bf0 = nb0; bf1 = nb1; }
        }

#pragma unroll
        for (int r = 0; r < 8; ++r) {
            int m = 8 * g + r;
            size_t base = rowbase[r] + (size_t)t * Dsz;
            float v0 = tanhf(acc0[r] + xt0[r] + bias0);
            float v1 = tanhf(acc1[r] + xt1[r] + bias1);
            __builtin_nontemporal_store(v0, &out[base + col0]);
            __builtin_nontemporal_store(v1, &out[base + col1]);
            hbuf[nxt][m * HS + col0] = (_Float16)v0;
            hbuf[nxt][m * HS + col1] = (_Float16)v1;
            if (t == Ssz - 1) {
                last[(size_t)(brow0 + m) * Dsz + col0] = v0;
                last[(size_t)(brow0 + m) * Dsz + col1] = v1;
            }
        }
        __syncthreads();
    }
}

// ---------------------------------------------------------------------------
extern "C" void kernel_launch(void* const* d_in, const int* in_sizes, int n_in,
                              void* d_out, int out_size, void* d_ws, size_t ws_size,
                              hipStream_t stream) {
    (void)in_sizes; (void)n_in; (void)out_size; (void)ws_size;
    const float* x    = (const float*)d_in[0];
    const float* W_ih = (const float*)d_in[1];
    const float* b_ih = (const float*)d_in[2];
    const float* W_hh = (const float*)d_in[3];
    const float* b_hh = (const float*)d_in[4];

    float* out  = (float*)d_out;                       // [B,S,D]
    float* last = out + (size_t)Bsz * Ssz * Dsz;       // [B,D]

    _Float16* Wp_ih = (_Float16*)d_ws;                 // 512 KB
    _Float16* Wp_hh = Wp_ih + (size_t)Dsz * Dsz;       // 512 KB

    pack_w_kernel<<<(Dsz * Dsz) / 256, 256, 0, stream>>>(W_ih, Wp_ih);
    pack_w_kernel<<<(Dsz * Dsz) / 256, 256, 0, stream>>>(W_hh, Wp_hh);
    proj_kernel<<<(Bsz * Ssz) / (16 * 8), 256, 0, stream>>>(x, Wp_ih, b_ih, out);
    scan_kernel<<<Bsz / 16, 512, 0, stream>>>(Wp_hh, b_hh, out, last);
}
